// GCN_with_MLP_11768210391288
// MI455X (gfx1250) — compile-verified, matrix operand order checked by reading, not run on hardware
//
#include <hip/hip_runtime.h>
#include <hip/hip_bf16.h>

#define FEAT 128
#define EPSV 1e-5f

typedef __attribute__((ext_vector_type(16))) __bf16 v16bf;
typedef __attribute__((ext_vector_type(8)))  float  v8f;

__device__ __forceinline__ unsigned short f2bf(float f) {
  unsigned u = __float_as_uint(f);
  u += 0x7FFFu + ((u >> 16) & 1u);   // round-to-nearest-even
  return (unsigned short)(u >> 16);
}

// ---------------- degree / dinv ----------------
__global__ void zero_f32(float* __restrict__ p, int n) {
  int i = blockIdx.x * blockDim.x + threadIdx.x;
  if (i < n) p[i] = 0.0f;
}

__global__ void deg_accum(const int* __restrict__ dst, float* __restrict__ deg, int E) {
  int i = blockIdx.x * blockDim.x + threadIdx.x;
  if (i < E) atomicAdd(&deg[dst[i]], 1.0f);
}

__global__ void deg_rsqrt(float* __restrict__ deg, int N) {
  int i = blockIdx.x * blockDim.x + threadIdx.x;
  if (i < N) deg[i] = rsqrtf(deg[i] + 1.0f);
}

// ---------------- WMMA GEMM: out[N x 128] = bf16(A) @ bf16(W) (+bias,+relu) ----
// A is [N x K] fp32, logically concat(A0[N x 128], A1[N x 128]) when K==256.
// W is [K x 128] fp32 row-major. Block: 256 thr = 8 waves; 64 rows x 128 cols
// per block. Each wave owns one 16-col tile across 4 row strips (4 accums),
// so each staged W slice and each B fragment is reused 4x. K consumed in
// 128-wide LDS-staged halves.
template<int K, bool RELU, bool BIAS>
__global__ __launch_bounds__(256) void gemm_wmma(
    const float* __restrict__ A0, const float* __restrict__ A1,
    const float* __restrict__ W,  const float* __restrict__ bias,
    float* __restrict__ out, int n_rows)
{
  constexpr int STR = 136;  // 128 + 8 ushort padding: bank-conflict-free b128 DS loads
  __shared__ __align__(16) unsigned short sA[64 * STR];    // 64 x 128 A strip
  __shared__ __align__(16) unsigned short sW[128 * STR];   // transposed: sW[n*STR + k]

  const int t     = threadIdx.x;
  const int row0  = blockIdx.x * 64;
  const int wave  = t >> 5;
  const int lane  = t & 31;
  const int nloc  = lane & 15;           // A row / B col within 16x16 tile
  const int kbase = (lane >> 4) << 3;    // lanes 16-31 hold K+8 per ISA layout

  v8f acc[4];
#pragma unroll
  for (int rs = 0; rs < 4; ++rs) acc[rs] = v8f{0.f,0.f,0.f,0.f,0.f,0.f,0.f,0.f};

  union Frag { v16bf v; uint4 q[2]; };

  for (int h = 0; h < K / 128; ++h) {
    // stage 64 x 128 slice of A (fp32 -> bf16), row-clamped for the tail block
    for (int idx = t; idx < 64 * 128; idx += 256) {
      int r = idx >> 7, c = idx & 127;
      int gr = row0 + r; if (gr >= n_rows) gr = n_rows - 1;
      int gc = h * 128 + c;
      float v = (gc < 128) ? A0[(size_t)gr * 128 + gc]
                           : A1[(size_t)gr * 128 + (gc - 128)];
      sA[r * STR + c] = f2bf(v);
    }
    // stage 128 x 128 slice of W, transposed into sW[n][k]
    for (int idx = t; idx < 128 * 128; idx += 256) {
      int k = idx >> 7, n = idx & 127;
      sW[n * STR + k] = f2bf(W[(size_t)(h * 128 + k) * 128 + n]);
    }
    __syncthreads();

    const unsigned short* aBase = &sA[nloc * STR + kbase];
    const unsigned short* bBase = &sW[(wave * 16 + nloc) * STR + kbase];
#pragma unroll
    for (int kk = 0; kk < 128; kk += 32) {
      Frag b;
      b.q[0] = *(const uint4*)(bBase + kk);        // K = kbase .. kbase+7
      b.q[1] = *(const uint4*)(bBase + kk + 16);   // K = kbase+16 .. kbase+23
#pragma unroll
      for (int rs = 0; rs < 4; ++rs) {
        Frag a;
        const unsigned short* ap = aBase + rs * (16 * STR) + kk;
        a.q[0] = *(const uint4*)ap;
        a.q[1] = *(const uint4*)(ap + 16);
        acc[rs] = __builtin_amdgcn_wmma_f32_16x16x32_bf16(
            false, a.v, false, b.v, (short)0, acc[rs], false, false);
      }
    }
    __syncthreads();
  }

  // C/D layout: VGPR r -> row rbase+r, col = lane%16 (lanes>=16: rows +8)
  const int col   = wave * 16 + nloc;
  const int rbase = (lane >> 4) << 3;
  float bv = BIAS ? bias[col] : 0.0f;

  if (row0 + 64 <= n_rows) {
    // full tile: straight-line stores, no per-element predication
#pragma unroll
    for (int rs = 0; rs < 4; ++rs) {
#pragma unroll
      for (int r = 0; r < 8; ++r) {
        float v = acc[rs][r] + bv;
        if (RELU) v = fmaxf(v, 0.0f);
        out[(size_t)(row0 + rs * 16 + rbase + r) * 128 + col] = v;
      }
    }
  } else {
#pragma unroll
    for (int rs = 0; rs < 4; ++rs) {
#pragma unroll
      for (int r = 0; r < 8; ++r) {
        int grow = row0 + rs * 16 + rbase + r;
        if (grow < n_rows) {
          float v = acc[rs][r] + bv;
          if (RELU) v = fmaxf(v, 0.0f);
          out[(size_t)grow * 128 + col] = v;
        }
      }
    }
  }
}

// ---------------- GCN aggregation ----------------
// agg[n,c] = xw[n,c]*dinv[n]^2 + b[c]     (self-loop term + bias)
__global__ void self_init(const float* __restrict__ xw, const float* __restrict__ dinv,
                          const float* __restrict__ bias, float* __restrict__ agg, int N) {
  int i = blockIdx.x * blockDim.x + threadIdx.x;
  if (i < N * FEAT) {
    int n = i >> 7, c = i & 127;
    float di = dinv[n];
    agg[i] = xw[i] * di * di + bias[c];
  }
}

// one wave32 per edge: gather 128 floats (float4/lane), scale, scatter-add
__global__ __launch_bounds__(256) void edge_scatter(
    const float* __restrict__ hsrc, const float* __restrict__ dinv,
    const int* __restrict__ src, const int* __restrict__ dst,
    float* __restrict__ agg, int E)
{
  int gid  = blockIdx.x * blockDim.x + threadIdx.x;
  int e    = gid >> 5;
  int lane = gid & 31;
  if (e >= E) return;
  int s = src[e], d = dst[e];
  float coef = dinv[s] * dinv[d];
  const float4 hv = *(const float4*)(hsrc + (size_t)s * 128 + lane * 4);
  float* ap = agg + (size_t)d * 128 + lane * 4;
  atomicAdd(ap + 0, hv.x * coef);
  atomicAdd(ap + 1, hv.y * coef);
  atomicAdd(ap + 2, hv.z * coef);
  atomicAdd(ap + 3, hv.w * coef);
}

// ---------------- LayerNorm + ReLU: one wave32 per row ----------------
__global__ __launch_bounds__(256) void ln_relu(
    const float* __restrict__ in, const float* __restrict__ g,
    const float* __restrict__ be, float* __restrict__ out, int N)
{
  int gid  = blockIdx.x * blockDim.x + threadIdx.x;
  int row  = gid >> 5;
  int lane = gid & 31;
  if (row >= N) return;
  const float4 v = *(const float4*)(in + (size_t)row * 128 + lane * 4);
  float s  = v.x + v.y + v.z + v.w;
  float ss = v.x * v.x + v.y * v.y + v.z * v.z + v.w * v.w;
#pragma unroll
  for (int m = 16; m >= 1; m >>= 1) {
    s  += __shfl_xor(s,  m, 32);
    ss += __shfl_xor(ss, m, 32);
  }
  float mean = s  * (1.0f / 128.0f);
  float var  = ss * (1.0f / 128.0f) - mean * mean;
  float rstd = rsqrtf(var + EPSV);
  int c = lane * 4;
  float4 o;
  o.x = fmaxf((v.x - mean) * rstd * g[c + 0] + be[c + 0], 0.0f);
  o.y = fmaxf((v.y - mean) * rstd * g[c + 1] + be[c + 1], 0.0f);
  o.z = fmaxf((v.z - mean) * rstd * g[c + 2] + be[c + 2], 0.0f);
  o.w = fmaxf((v.w - mean) * rstd * g[c + 3] + be[c + 3], 0.0f);
  *(float4*)(out + (size_t)row * 128 + c) = o;
}

// ---------------- final 128 -> 2 projection: one wave32 per row -------
__global__ __launch_bounds__(256) void mlp_out(
    const float* __restrict__ h, const float* __restrict__ Wm2,
    const float* __restrict__ bm2, float* __restrict__ out, int N)
{
  int gid  = blockIdx.x * blockDim.x + threadIdx.x;
  int row  = gid >> 5;
  int lane = gid & 31;
  if (row >= N) return;
  const float4 hv = *(const float4*)(h + (size_t)row * 128 + lane * 4);
  const float4 w0 = *(const float4*)(Wm2 + lane * 8);       // [k,0],[k,1],[k+1,0],[k+1,1]
  const float4 w1 = *(const float4*)(Wm2 + lane * 8 + 4);   // [k+2,..],[k+3,..]
  float s0 = hv.x * w0.x + hv.y * w0.z + hv.z * w1.x + hv.w * w1.z;
  float s1 = hv.x * w0.y + hv.y * w0.w + hv.z * w1.y + hv.w * w1.w;
#pragma unroll
  for (int m = 16; m >= 1; m >>= 1) {
    s0 += __shfl_xor(s0, m, 32);
    s1 += __shfl_xor(s1, m, 32);
  }
  if (lane == 0) {
    out[(size_t)row * 2 + 0] = s0 + bm2[0];
    out[(size_t)row * 2 + 1] = s1 + bm2[1];
  }
}

extern "C" void kernel_launch(void* const* d_in, const int* in_sizes, int n_in,
                              void* d_out, int out_size, void* d_ws, size_t ws_size,
                              hipStream_t stream)
{
  const float* x   = (const float*)d_in[0];
  const int*   ei  = (const int*)  d_in[1];
  const float* W1  = (const float*)d_in[2];
  const float* b1  = (const float*)d_in[3];
  const float* g1  = (const float*)d_in[4];
  const float* be1 = (const float*)d_in[5];
  const float* W2  = (const float*)d_in[6];
  const float* b2  = (const float*)d_in[7];
  const float* g2  = (const float*)d_in[8];
  const float* be2 = (const float*)d_in[9];
  const float* Wm1 = (const float*)d_in[10];
  const float* bm1 = (const float*)d_in[11];
  const float* Wm2 = (const float*)d_in[12];
  const float* bm2 = (const float*)d_in[13];

  const int N = in_sizes[0] / FEAT;
  const int E = in_sizes[1] / 2;
  const int* src = ei;          // edge_index[0]
  const int* dst = ei + E;      // edge_index[1]

  // workspace layout: dinv[N] | bufA | bufB | bufC  (each N*128 fp32)
  float* dinv = (float*)d_ws;
  float* bufA = dinv + N;
  float* bufB = bufA + (size_t)N * FEAT;
  float* bufC = bufB + (size_t)N * FEAT;

  const int bN  = (N + 255) / 256;
  const int bE  = (E + 255) / 256;
  const int bNE = (int)(((size_t)N * FEAT + 255) / 256);
  const int bG  = (N + 63) / 64;           // GEMM blocks (64 rows each)
  const int bW  = (E + 7) / 8;             // edge waves: 8 per 256-thr block
  const int bR  = (N + 7) / 8;             // row waves: 8 per block

  // degree -> dinv (re-zeroed every launch for determinism under graph replay)
  zero_f32 <<<bN, 256, 0, stream>>>(dinv, N);
  deg_accum<<<bE, 256, 0, stream>>>(dst, dinv, E);
  deg_rsqrt<<<bN, 256, 0, stream>>>(dinv, N);

  // ---- GCN layer 1 ----
  gemm_wmma<128, false, false><<<bG, 256, 0, stream>>>(x, nullptr, W1, nullptr, bufA, N);
  self_init   <<<bNE, 256, 0, stream>>>(bufA, dinv, b1, bufB, N);
  edge_scatter<<<bW,  256, 0, stream>>>(bufA, dinv, src, dst, bufB, E);
  ln_relu     <<<bR,  256, 0, stream>>>(bufB, g1, be1, bufA, N);

  // ---- GCN layer 2 ----
  gemm_wmma<128, false, false><<<bG, 256, 0, stream>>>(bufA, nullptr, W2, nullptr, bufB, N);
  self_init   <<<bNE, 256, 0, stream>>>(bufB, dinv, b2, bufC, N);
  edge_scatter<<<bW,  256, 0, stream>>>(bufB, dinv, src, dst, bufC, E);
  ln_relu     <<<bR,  256, 0, stream>>>(bufC, g2, be2, bufB, N);

  // ---- MLP: relu(concat(h2, x) @ Wm1 + bm1) then @ Wm2 + bm2 ----
  gemm_wmma<256, true, true><<<bG, 256, 0, stream>>>(bufB, x, Wm1, bm1, bufA, N);
  mlp_out<<<bR, 256, 0, stream>>>(bufA, Wm2, bm2, (float*)d_out, N);
}
